// TransformerModel_68547678044207
// MI455X (gfx1250) — matrix operand order repaired and checked by reading, not compiled, not run on hardware
//
#include <hip/hip_runtime.h>
#include <hip/hip_bf16.h>
#include <stdint.h>

typedef __attribute__((ext_vector_type(16))) __bf16 v16bf;
typedef __attribute__((ext_vector_type(8)))  float  v8f;

#define Bn   16
#define Sn   1024
#define CCn  8
#define CFn  8
#define En   64
#define Hn   1024
#define NHn  16
#define DHn  64
#define TOKS (Bn*Sn)
#define EPSf 1e-5f

// ---------- helpers ----------
__device__ __forceinline__ unsigned short f2bf(float f) {
  unsigned u = __builtin_bit_cast(unsigned, f);
  unsigned r = (u + 0x7FFFu + ((u >> 16) & 1u)) >> 16;   // RNE
  return (unsigned short)r;
}
__device__ __forceinline__ float bf2f(unsigned short h) {
  unsigned u = ((unsigned)h) << 16;
  return __builtin_bit_cast(float, u);
}
__device__ __forceinline__ v8f wmma_bf16(v16bf a, v16bf b, v8f c) {
  return __builtin_amdgcn_wmma_f32_16x16x32_bf16(false, a, false, b, (short)0, c, false, false);
}

// ---- WMMA-fragment-native storage ----
// A 16x32 bf16 tile and B 32x16 bf16 tile are both stored as
// [tile][lane 0..31][16 contiguous bf16]  (512 bf16 = 1KB per tile),
// so a fragment load is one aligned 32-byte per-lane read.

// Offset (in ushorts) of A element (row m, k within 16 x K fragment buffer).
// Inverse of the ISA 7.12.2 16-bit A layout:
//   lanes 0-15: m, K pairs at VGPR j: K = 2j + (j<4 ? hi*8 : 8+hi*8) + half
__device__ __forceinline__ int a_off(int m, int k) {
  int kt   = k >> 5;
  int kk   = k & 31;
  int hi   = (kk >> 3) & 1;            // kk in [8,16)|[24,32) -> upper half-lanes
  int grp  = kk >> 4;                  // 0: j<4, 1: j>=4
  int j    = ((kk & 7) >> 1) + grp * 4;
  int e    = 2 * j + (kk & 1);
  return ((kt << 5) + (hi << 4) + m) * 16 + e;
}

// Load A fragment (tile kt) from fragment-layout LDS buffer.
__device__ __forceinline__ v16bf load_af(const unsigned short* base, int kt) {
  int lane = threadIdx.x & 31;
  return *(const v16bf*)(base + (((kt << 5) + lane) << 4));
}

// Load B fragment (K-tile kt, N-tile nt) from fragment-layout global weights.
__device__ __forceinline__ v16bf load_bf(const unsigned short* base, int ntiles_n, int kt, int nt) {
  int lane = threadIdx.x & 31;
  return *(const v16bf*)(base + (((kt * ntiles_n + nt) << 5) + lane) * 16);
}
__device__ __forceinline__ const void* b_tile_ptr(const unsigned short* base, int ntiles_n, int kt, int nt) {
  int lane = threadIdx.x & 31;
  return (const void*)(base + (((kt * ntiles_n + nt) << 5) + lane) * 16);
}

// ---------- kernel 1: BatchNorm batch stats over (B*S, CF) ----------
__global__ __launch_bounds__(256)
void bn_stats_kernel(const float* __restrict__ cont_x,
                     const float* __restrict__ bn_g, const float* __restrict__ bn_b,
                     float* __restrict__ bnScale, float* __restrict__ bnShift) {
  __shared__ float S1[256 * 8];
  __shared__ float S2[256 * 8];
  const int tid = threadIdx.x;
  float s1[8] = {0,0,0,0,0,0,0,0}, s2[8] = {0,0,0,0,0,0,0,0};
  for (int r = tid; r < TOKS; r += 256) {
#pragma unroll
    for (int f = 0; f < CFn; ++f) {
      float v = cont_x[r * CFn + f];
      s1[f] += v; s2[f] += v * v;
    }
  }
#pragma unroll
  for (int f = 0; f < CFn; ++f) { S1[tid * 8 + f] = s1[f]; S2[tid * 8 + f] = s2[f]; }
  __syncthreads();
  for (int str = 128; str > 0; str >>= 1) {
    if (tid < str) {
#pragma unroll
      for (int f = 0; f < CFn; ++f) {
        S1[tid * 8 + f] += S1[(tid + str) * 8 + f];
        S2[tid * 8 + f] += S2[(tid + str) * 8 + f];
      }
    }
    __syncthreads();
  }
  if (tid < CFn) {
    float mu  = S1[tid] * (1.f / TOKS);
    float var = S2[tid] * (1.f / TOKS) - mu * mu;
    float sc  = bn_g[tid] * rsqrtf(var + EPSf);
    bnScale[tid] = sc;
    bnShift[tid] = bn_b[tid] - mu * sc;
  }
}

// ---------- kernel 2: f32 [K x N] -> bf16 fragment-tile layout ----------
// dst element o = ((kt*(N/16)+nt)*32 + lane)*16 + e
//   holds src[(kt*32 + (e>=8?16:0) + (lane>=16?8:0) + (e&7)) * N + nt*16 + (lane&15)]
__global__ __launch_bounds__(256)
void cvt_tile_kernel(const float* __restrict__ src, unsigned short* __restrict__ dst,
                     int K, int N) {
  const int ntn = N >> 4;
  const int total = K * N;
  for (int o = blockIdx.x * 256 + threadIdx.x; o < total; o += gridDim.x * 256) {
    int e    = o & 15;
    int lane = (o >> 4) & 31;
    int t    = o >> 9;
    int nt   = t % ntn;
    int kt   = t / ntn;
    int k    = (kt << 5) + ((e >> 3) << 4) + ((lane >> 4) << 3) + (e & 7);
    int n    = (nt << 4) + (lane & 15);
    dst[o] = f2bf(src[k * N + n]);
  }
}

// ---------- kernel 3: per-token MLP + K/V projection (WMMA) ----------
// One block = 16 tokens, 256 threads = 8 waves; wave w owns N-tiles [w*8, w*8+8).
__global__ __launch_bounds__(256)
void token_kernel(const int* __restrict__ cate_x, const float* __restrict__ cont_x,
                  const float* __restrict__ emb_table,
                  const float* __restrict__ bnScale, const float* __restrict__ bnShift,
                  const float* __restrict__ cont_W, const float* __restrict__ cont_b,
                  const float* __restrict__ cont_ln_g, const float* __restrict__ cont_ln_b,
                  const unsigned short* __restrict__ cateWb, const float* __restrict__ cate_b,
                  const float* __restrict__ cate_ln_g, const float* __restrict__ cate_ln_b,
                  const unsigned short* __restrict__ combWb, const float* __restrict__ comb_b,
                  const float* __restrict__ comb_ln_g, const float* __restrict__ comb_ln_b,
                  const unsigned short* __restrict__ WkB, const float* __restrict__ bk,
                  const unsigned short* __restrict__ WvB, const float* __restrict__ bv,
                  unsigned short* __restrict__ Kbuf, unsigned short* __restrict__ Vbuf,
                  float* __restrict__ xlast) {
  __shared__ __align__(32) unsigned short buf[16 * 1024];   // 32KB, A-fragment layout
  __shared__ __align__(32) unsigned short acat[16 * 512];   // 16KB, A-fragment layout
  __shared__ float rs[16], rq[16];                          // row stats

  const int tid = threadIdx.x;
  const int w = tid >> 5, lane = tid & 31;
  const int hi = lane >> 4, nl = lane & 15;
  const int t0 = blockIdx.x * 16;
  const v8f vzero = {0.f,0.f,0.f,0.f,0.f,0.f,0.f,0.f};

  // --- phase 0: gather embeddings into A-fragment layout ---
  if (tid < 16) { rs[tid] = 0.f; rq[tid] = 0.f; }
#pragma unroll
  for (int i = 0; i < 32; ++i) {
    int g = tid + 256 * i;              // 0..8191
    int m = g >> 9, kk = g & 511;
    int c = kk >> 6, d = kk & 63;
    int idx = cate_x[(t0 + m) * CCn + c];
    acat[a_off(m, kk)] = f2bf(emb_table[idx * En + d]);
  }
  __syncthreads();                                           // B1

  // --- phase 1: cate GEMM [16x512]x[512x1024] + LN + relu -> buf ---
  v8f acc[8];
#pragma unroll
  for (int s = 0; s < 8; ++s) acc[s] = vzero;
  for (int kk = 0; kk < 16; ++kk) {
    v16bf a = load_af(acat, kk);
    if (kk + 1 < 16) __builtin_prefetch(b_tile_ptr(cateWb, 64, kk + 1, w * 8), 0, 1);
#pragma unroll
    for (int s = 0; s < 8; ++s) {
      v16bf b = load_bf(cateWb, 64, kk, w * 8 + s);
      acc[s] = wmma_bf16(a, b, acc[s]);
    }
  }
  {
    float p1[8] = {0,0,0,0,0,0,0,0}, p2[8] = {0,0,0,0,0,0,0,0};
#pragma unroll
    for (int s = 0; s < 8; ++s) {
      int n = w * 128 + s * 16 + nl;
      float bia = cate_b[n];
#pragma unroll
      for (int v = 0; v < 8; ++v) {
        float t = acc[s][v] + bia; acc[s][v] = t;
        p1[v] += t; p2[v] += t * t;
      }
    }
#pragma unroll
    for (int v = 0; v < 8; ++v) {
      atomicAdd(&rs[hi * 8 + v], p1[v]);
      atomicAdd(&rq[hi * 8 + v], p2[v]);
    }
  }
  __syncthreads();                                           // B2
  {
    float mn[8], rv[8];
#pragma unroll
    for (int v = 0; v < 8; ++v) {
      int r = hi * 8 + v;
      float m = rs[r] * (1.f / 1024.f);
      float var = rq[r] * (1.f / 1024.f) - m * m;
      mn[v] = m; rv[v] = rsqrtf(var + EPSf);
    }
#pragma unroll
    for (int s = 0; s < 8; ++s) {
      int n = w * 128 + s * 16 + nl;
      float g = cate_ln_g[n], be = cate_ln_b[n];
#pragma unroll
      for (int v = 0; v < 8; ++v) {
        float t = (acc[s][v] - mn[v]) * rv[v] * g + be;
        buf[a_off(hi * 8 + v, n)] = f2bf(fmaxf(t, 0.f));
      }
    }
  }
  __syncthreads();                                           // B3
  if (tid < 16) { rs[tid] = 0.f; rq[tid] = 0.f; }

  // --- phase 2a: comb GEMM first K-half (cate part of seq) ---
  v8f acc3[8];
#pragma unroll
  for (int s = 0; s < 8; ++s) acc3[s] = vzero;
  for (int kk = 0; kk < 32; ++kk) {
    v16bf a = load_af(buf, kk);
    if (kk + 1 < 32) __builtin_prefetch(b_tile_ptr(combWb, 64, kk + 1, w * 8), 0, 1);
#pragma unroll
    for (int s = 0; s < 8; ++s) {
      v16bf b = load_bf(combWb, 64, kk, w * 8 + s);
      acc3[s] = wmma_bf16(a, b, acc3[s]);
    }
  }
  __syncthreads();                                           // B4

  // --- phase 2b: cont path (BN + GEMV + LN + relu) -> buf (overwrites) ---
  {
    int m = tid >> 4, c16 = tid & 15;
    int tg = t0 + m;
    float xn[8];
#pragma unroll
    for (int f = 0; f < CFn; ++f)
      xn[f] = cont_x[tg * CFn + f] * bnScale[f] + bnShift[f];
    float s1 = 0.f, s2 = 0.f;
    for (int i = 0; i < 64; ++i) {
      int n = c16 * 64 + i;
      float pre = cont_b[n];
#pragma unroll
      for (int f = 0; f < CFn; ++f) pre += xn[f] * cont_W[f * Hn + n];
      s1 += pre; s2 += pre * pre;
    }
    atomicAdd(&rs[m], s1); atomicAdd(&rq[m], s2);
    __syncthreads();                                         // B5
    float mm = rs[m] * (1.f / 1024.f);
    float var = rq[m] * (1.f / 1024.f) - mm * mm;
    float rstd = rsqrtf(var + EPSf);
    for (int i = 0; i < 64; ++i) {
      int n = c16 * 64 + i;
      float pre = cont_b[n];
#pragma unroll
      for (int f = 0; f < CFn; ++f) pre += xn[f] * cont_W[f * Hn + n];
      float t = (pre - mm) * rstd * cont_ln_g[n] + cont_ln_b[n];
      buf[a_off(m, n)] = f2bf(fmaxf(t, 0.f));
    }
  }
  __syncthreads();                                           // B6
  if (tid < 16) { rs[tid] = 0.f; rq[tid] = 0.f; }

  // --- phase 2c: comb GEMM second K-half, bias, LN -> x tile (buf) ---
  for (int kk = 0; kk < 32; ++kk) {
    v16bf a = load_af(buf, kk);
    if (kk + 1 < 32) __builtin_prefetch(b_tile_ptr(combWb, 64, 32 + kk + 1, w * 8), 0, 1);
#pragma unroll
    for (int s = 0; s < 8; ++s) {
      v16bf b = load_bf(combWb, 64, 32 + kk, w * 8 + s);
      acc3[s] = wmma_bf16(a, b, acc3[s]);
    }
  }
  __syncthreads();                                           // B7
  {
    float p1[8] = {0,0,0,0,0,0,0,0}, p2[8] = {0,0,0,0,0,0,0,0};
#pragma unroll
    for (int s = 0; s < 8; ++s) {
      int n = w * 128 + s * 16 + nl;
      float bia = comb_b[n];
#pragma unroll
      for (int v = 0; v < 8; ++v) {
        float t = acc3[s][v] + bia; acc3[s][v] = t;
        p1[v] += t; p2[v] += t * t;
      }
    }
#pragma unroll
    for (int v = 0; v < 8; ++v) {
      atomicAdd(&rs[hi * 8 + v], p1[v]);
      atomicAdd(&rq[hi * 8 + v], p2[v]);
    }
  }
  __syncthreads();                                           // B8
  {
    const bool lastTile = ((t0 & (Sn - 1)) == Sn - 16);
    const int bIdx = t0 >> 10;
    float mn[8], rv[8];
#pragma unroll
    for (int v = 0; v < 8; ++v) {
      int r = hi * 8 + v;
      float m = rs[r] * (1.f / 1024.f);
      float var = rq[r] * (1.f / 1024.f) - m * m;
      mn[v] = m; rv[v] = rsqrtf(var + EPSf);
    }
#pragma unroll
    for (int s = 0; s < 8; ++s) {
      int n = w * 128 + s * 16 + nl;
      float g = comb_ln_g[n], be = comb_ln_b[n];
#pragma unroll
      for (int v = 0; v < 8; ++v) {
        int r = hi * 8 + v;
        float t = (acc3[s][v] - mn[v]) * rv[v] * g + be;
        buf[a_off(r, n)] = f2bf(t);
        if (lastTile && r == 15) xlast[bIdx * Hn + n] = t;  // row for Q
      }
    }
  }
  __syncthreads();                                           // B9

  // --- phase 3: K projection ---
  v8f ak[8];
#pragma unroll
  for (int s = 0; s < 8; ++s) ak[s] = vzero;
  for (int kk = 0; kk < 32; ++kk) {
    v16bf a = load_af(buf, kk);
    if (kk + 1 < 32) __builtin_prefetch(b_tile_ptr(WkB, 64, kk + 1, w * 8), 0, 1);
#pragma unroll
    for (int s = 0; s < 8; ++s) {
      v16bf b = load_bf(WkB, 64, kk, w * 8 + s);
      ak[s] = wmma_bf16(a, b, ak[s]);
    }
  }
#pragma unroll
  for (int s = 0; s < 8; ++s) {
    int n = w * 128 + s * 16 + nl;
    float bia = bk[n];
#pragma unroll
    for (int v = 0; v < 8; ++v)
      Kbuf[(size_t)(t0 + hi * 8 + v) * Hn + n] = f2bf(ak[s][v] + bia);
  }

  // --- phase 4: V projection ---
  v8f av[8];
#pragma unroll
  for (int s = 0; s < 8; ++s) av[s] = vzero;
  for (int kk = 0; kk < 32; ++kk) {
    v16bf a = load_af(buf, kk);
    if (kk + 1 < 32) __builtin_prefetch(b_tile_ptr(WvB, 64, kk + 1, w * 8), 0, 1);
#pragma unroll
    for (int s = 0; s < 8; ++s) {
      v16bf b = load_bf(WvB, 64, kk, w * 8 + s);
      av[s] = wmma_bf16(a, b, av[s]);
    }
  }
#pragma unroll
  for (int s = 0; s < 8; ++s) {
    int n = w * 128 + s * 16 + nl;
    float bia = bv[n];
#pragma unroll
    for (int v = 0; v < 8; ++v)
      Vbuf[(size_t)(t0 + hi * 8 + v) * Hn + n] = f2bf(av[s][v] + bia);
  }
}

// ---------- kernel 4: attention for query position S-1, one wave per (b,head) ----------
__global__ __launch_bounds__(256)
void attn_kernel(const unsigned short* __restrict__ Kbuf, const unsigned short* __restrict__ Vbuf,
                 const float* __restrict__ xlast, const float* __restrict__ Wq,
                 const float* __restrict__ bq, const int* __restrict__ mask,
                 float* __restrict__ attnout) {
  __shared__ float qsh[8][64];
  __shared__ float psh[8][1024];
  const int tid = threadIdx.x;
  const int w = tid >> 5, lane = tid & 31;
  const int task = blockIdx.x * 8 + w;
  const int b = task >> 4, h = task & 15;

  // Q row: q[d] = xlast[b] . Wq[:, h*64+d] + bq
#pragma unroll
  for (int rep = 0; rep < 2; ++rep) {
    int d = lane + rep * 32;
    int col = h * 64 + d;
    float acc = bq[col];
    for (int j = 0; j < Hn; ++j)
      acc += xlast[b * Hn + j] * Wq[j * Hn + col];
    qsh[w][d] = acc;
  }
  __syncthreads();

  const int mk = mask[b * Sn + (Sn - 1)];
  float sc[32];
  const unsigned short* Kb = Kbuf + ((size_t)b * Sn) * Hn + h * 64;
  for (int i = 0; i < 32; ++i) {
    const unsigned short* kr = Kb + (size_t)(i * 32 + lane) * Hn;
    float s = 0.f;
#pragma unroll
    for (int d = 0; d < DHn; ++d) s += qsh[w][d] * bf2f(kr[d]);
    sc[i] = s * 0.125f;                 // 1/sqrt(64)
  }
  if (mk == 0) {
    // all scores replaced by -1e7 -> softmax uniform
#pragma unroll
    for (int i = 0; i < 32; ++i) psh[w][i * 32 + lane] = 1.f / 1024.f;
  } else {
    float mx = -3.4e38f;
#pragma unroll
    for (int i = 0; i < 32; ++i) mx = fmaxf(mx, sc[i]);
    for (int off = 16; off; off >>= 1) mx = fmaxf(mx, __shfl_xor(mx, off, 32));
    float sm = 0.f;
#pragma unroll
    for (int i = 0; i < 32; ++i) { sc[i] = __expf(sc[i] - mx); sm += sc[i]; }
    for (int off = 16; off; off >>= 1) sm += __shfl_xor(sm, off, 32);
    float inv = 1.f / sm;
#pragma unroll
    for (int i = 0; i < 32; ++i) psh[w][i * 32 + lane] = sc[i] * inv;
  }
  __syncthreads();

  const unsigned short* Vb = Vbuf + ((size_t)b * Sn) * Hn + h * 64;
#pragma unroll
  for (int rep = 0; rep < 2; ++rep) {
    int d = lane + rep * 32;
    float o = 0.f;
    for (int k = 0; k < Sn; ++k)
      o += psh[w][k] * bf2f(Vb[(size_t)k * Hn + d]);
    attnout[b * Hn + h * 64 + d] = o;
  }
}

// ---------- kernel 5: final dot with Wo[:,H-1] ----------
__global__ __launch_bounds__(256)
void final_kernel(const float* __restrict__ attnout, const float* __restrict__ Wo,
                  const float* __restrict__ bo, float* __restrict__ out) {
  __shared__ float red[256];
  const int b = blockIdx.x, tid = threadIdx.x;
  float s = 0.f;
  for (int i = tid; i < Hn; i += 256)
    s += attnout[b * Hn + i] * Wo[i * Hn + (Hn - 1)];
  red[tid] = s;
  __syncthreads();
  for (int str = 128; str > 0; str >>= 1) {
    if (tid < str) red[tid] += red[tid + str];
    __syncthreads();
  }
  if (tid == 0) out[b] = red[0] + bo[Hn - 1];
}

// ---------- host ----------
extern "C" void kernel_launch(void* const* d_in, const int* in_sizes, int n_in,
                              void* d_out, int out_size, void* d_ws, size_t ws_size,
                              hipStream_t stream) {
  (void)in_sizes; (void)n_in; (void)out_size; (void)ws_size;
  const int*   cate_x    = (const int*)  d_in[0];
  const float* cont_x    = (const float*)d_in[1];
  const int*   mask      = (const int*)  d_in[2];
  const float* emb       = (const float*)d_in[3];
  const float* cate_W    = (const float*)d_in[4];
  const float* cate_b    = (const float*)d_in[5];
  const float* cate_ln_g = (const float*)d_in[6];
  const float* cate_ln_b = (const float*)d_in[7];
  const float* bn_g      = (const float*)d_in[8];
  const float* bn_b      = (const float*)d_in[9];
  const float* cont_W    = (const float*)d_in[10];
  const float* cont_b    = (const float*)d_in[11];
  const float* cont_ln_g = (const float*)d_in[12];
  const float* cont_ln_b = (const float*)d_in[13];
  const float* comb_W    = (const float*)d_in[14];
  const float* comb_b    = (const float*)d_in[15];
  const float* comb_ln_g = (const float*)d_in[16];
  const float* comb_ln_b = (const float*)d_in[17];
  const float* Wq = (const float*)d_in[18];
  const float* bq = (const float*)d_in[19];
  const float* Wk = (const float*)d_in[20];
  const float* bk = (const float*)d_in[21];
  const float* Wv = (const float*)d_in[22];
  const float* bv = (const float*)d_in[23];
  const float* Wo = (const float*)d_in[24];
  const float* bo = (const float*)d_in[25];

  char* ws = (char*)d_ws;
  float* bnScale = (float*)(ws + 0);                         // 8 f32
  float* bnShift = (float*)(ws + 32);                        // 8 f32
  float* xlast   = (float*)(ws + 256);                       // 16x1024 f32
  float* attnout = (float*)(ws + 256 + 65536);               // 16x1024 f32
  unsigned short* cateWb = (unsigned short*)(ws + 256 + 131072);
  unsigned short* combWb = cateWb + 512 * 1024;
  unsigned short* WkB    = combWb + 2048 * 1024;
  unsigned short* WvB    = WkB    + 1024 * 1024;
  unsigned short* Kbuf   = WvB    + 1024 * 1024;             // 16384x1024 bf16
  unsigned short* Vbuf   = Kbuf   + (size_t)TOKS * Hn;       // 16384x1024 bf16

  bn_stats_kernel<<<1, 256, 0, stream>>>(cont_x, bn_g, bn_b, bnScale, bnShift);
  cvt_tile_kernel<<<1024, 256, 0, stream>>>(cate_W, cateWb, 512, 1024);
  cvt_tile_kernel<<<2048, 256, 0, stream>>>(comb_W, combWb, 2048, 1024);
  cvt_tile_kernel<<<1024, 256, 0, stream>>>(Wk, WkB, 1024, 1024);
  cvt_tile_kernel<<<1024, 256, 0, stream>>>(Wv, WvB, 1024, 1024);

  token_kernel<<<TOKS / 16, 256, 0, stream>>>(
      cate_x, cont_x, emb, bnScale, bnShift,
      cont_W, cont_b, cont_ln_g, cont_ln_b,
      cateWb, cate_b, cate_ln_g, cate_ln_b,
      combWb, comb_b, comb_ln_g, comb_ln_b,
      WkB, bk, WvB, bv, Kbuf, Vbuf, xlast);

  attn_kernel<<<(Bn * NHn) / 8, 256, 0, stream>>>(Kbuf, Vbuf, xlast, Wq, bq, mask, attnout);
  final_kernel<<<Bn, 256, 0, stream>>>(attnout, Wo, bo, (float*)d_out);
}
